// MambaResidualLayer_18640158065082
// MI455X (gfx1250) — compile-verified, matrix operand order checked by reading, not run on hardware
//
#include <hip/hip_runtime.h>
#include <hip/hip_bf16.h>
#include <math.h>

// ---------------------------------------------------------------------------
// Mamba2 residual stack for MI455X (gfx1250, wave32, WMMA).
// SSD chunked scan: chunk Q=64 turns the L=4096 recurrence into dense bf16
// WMMA matmuls + a cheap 64-step chunk-state scan.
// WMMA operand tiles are staged in LDS in *fragment order* (2x ds_load_b128
// per fragment); staging uses float4 global loads and 8-byte LDS stores
// wherever the K axis is the contiguous one.
// ---------------------------------------------------------------------------

typedef __attribute__((ext_vector_type(16))) __bf16 v16bf;
typedef __attribute__((ext_vector_type(8)))  float  v8f;

#define N_LAYERS 4
#define DMODEL   64
#define DINNER   128
#define DSTATE   128
#define NHEADS   2
#define HEADDIM  64
#define DINPROJ  514
#define ZXLD     528                 // padded row stride for zxbcdt
#define CONVDIM  384
#define BSZ      8
#define SEQ      4096
#define BL       (BSZ * SEQ)         // 32768 rows
#define NCHUNK   64
#define Q        64                  // chunk length
#define PS       (HEADDIM * DSTATE)  // 8192 state elems per (b,h)

__device__ __forceinline__ v8f zero8() {
  v8f z;
#pragma unroll
  for (int i = 0; i < 8; ++i) z[i] = 0.f;
  return z;
}

__device__ __forceinline__ v8f wmma_bf16(v16bf a, v16bf b, v8f c) {
  return __builtin_amdgcn_wmma_f32_16x16x32_bf16(false, a, false, b,
                                                 (short)0, c, false, false);
}

__device__ __forceinline__ float softplusf(float t) {
  return fmaxf(t, 0.f) + log1pf(__expf(-fabsf(t)));
}

// Fragment-order LDS index for a 16(row) x 32(K) WMMA operand block
// (ISA 7.12.2 16-bit layout): lane = (mn&15) + 16*((k>>3)&1),
// element j = (k&7) + 8*(k>>4). Works for A (mn=M) and N-major B (mn=N).
// Tiles are arrays of 512-element blocks: block = (mn>>4)*kblocks + (k>>5).
__device__ __forceinline__ int shuf_idx(int mn, int k, int kblocks) {
  int km = k & 31;
  return (((mn >> 4) * kblocks + (k >> 5)) << 9)
       + (((km >> 3) & 1) << 8)
       + ((mn & 15) << 4)
       + ((km & 7) | ((km >> 4) << 3));
}

// Load one lane's fragment (16 contiguous bf16 = 2 x b128) from a block.
__device__ __forceinline__ v16bf load_frag(const __bf16* block) {
  int lane = threadIdx.x & 31;
  const uint4* p = (const uint4*)(block + lane * 16);
  union { uint4 q[2]; v16bf v; } u;
  u.q[0] = p[0];
  u.q[1] = p[1];
  return u.v;
}

// Stage 4 consecutive K values (k % 4 == 0): one 8-byte LDS store.
__device__ __forceinline__ void stage4(__bf16* tile, int mn, int k, int kb,
                                       float4 v) {
  union { __bf16 h[4]; uint2 u; } p;
  p.h[0] = (__bf16)v.x; p.h[1] = (__bf16)v.y;
  p.h[2] = (__bf16)v.z; p.h[3] = (__bf16)v.w;
  *(uint2*)(tile + shuf_idx(mn, k, kb)) = p.u;
}

// Stage 4 consecutive rows (mn % 4 == 0, same k): stride-16 halfword stores.
__device__ __forceinline__ void stage4_rows(__bf16* tile, int mn, int k, int kb,
                                            float4 v) {
  int i0 = shuf_idx(mn, k, kb);
  tile[i0]      = (__bf16)v.x;
  tile[i0 + 16] = (__bf16)v.y;
  tile[i0 + 32] = (__bf16)v.z;
  tile[i0 + 48] = (__bf16)v.w;
}

__device__ __forceinline__ float4 ld4(const float* p) {
  return *(const float4*)p;
}

// ---------------------------------------------------------------------------
// Kernel 1: in_proj GEMM (32768 x 514 x 64) -> padded zxbcdt (ld 528).
// Block = 16 output rows, 8 waves sweep the 33 N-tiles. Branch-free staging.
// ---------------------------------------------------------------------------
__global__ void k_inproj(const float* __restrict__ x,
                         const float* __restrict__ iw,
                         float* __restrict__ zx) {
  __shared__ __align__(16) __bf16 sA[16 * 64];       // 1 mtile x 2 kblocks
  __shared__ __align__(16) __bf16 sB[8][16 * 64];    // per-wave, 1 ntile x 2 kb
  int tid = threadIdx.x;
  int m0 = blockIdx.x * 16;
  {  // 256 float4 packs, one per thread
    int r = tid >> 4, kp = (tid & 15) << 2;
    stage4(sA, r, kp, 2, ld4(x + (m0 + r) * 64 + kp));
  }
  __syncthreads();
  int wave = tid >> 5, lane = tid & 31;
  int col = lane & 15, rb = (lane >> 4) ? 8 : 0;
  __bf16* sBw = sB[wave];
  for (int nt = wave; nt < 33; nt += 8) {
    int n0 = nt * 16;
    for (int e = lane; e < 256; e += 32) {
      int j = e >> 4, kp = (e & 15) << 2;
      int gr = n0 + j;
      int grc = (gr < DINPROJ) ? gr : (DINPROJ - 1);
      float4 v = ld4(iw + grc * 64 + kp);
      v.x = (gr < DINPROJ) ? v.x : 0.f;
      v.y = (gr < DINPROJ) ? v.y : 0.f;
      v.z = (gr < DINPROJ) ? v.z : 0.f;
      v.w = (gr < DINPROJ) ? v.w : 0.f;
      stage4(sBw, j, kp, 2, v);
    }
    v8f acc = zero8();
#pragma unroll
    for (int ks = 0; ks < 2; ++ks) {
      v16bf a = load_frag(sA + (ks << 9));
      v16bf b = load_frag(sBw + (ks << 9));
      acc = wmma_bf16(a, b, acc);
    }
#pragma unroll
    for (int i = 0; i < 8; ++i)
      zx[(m0 + rb + i) * ZXLD + n0 + col] = acc[i];
  }
}

// ---------------------------------------------------------------------------
// Kernel 2: causal depthwise conv (width 4) + bias + SiLU, 4 channels/thread.
// Reads xBC straight out of zxbcdt (cols 128..511).
// ---------------------------------------------------------------------------
__global__ void k_conv(const float* __restrict__ zx,
                       const float* __restrict__ cw,
                       const float* __restrict__ cb,
                       float* __restrict__ xbca) {
  int idx = blockIdx.x * 256 + threadIdx.x;  // < BL*96
  int cp = (idx % 96) << 2;
  int bl = idx / 96;
  int l = bl % SEQ;
  float4 xs[4];
#pragma unroll
  for (int j = 0; j < 4; ++j) {
    int t = l - 3 + j;
    if (t >= 0) {
      xs[j] = ld4(zx + (bl + (t - l)) * ZXLD + DINNER + cp);
    } else {
      xs[j].x = xs[j].y = xs[j].z = xs[j].w = 0.f;
    }
  }
  float4 w0 = ld4(cw + (cp + 0) * 4);
  float4 w1 = ld4(cw + (cp + 1) * 4);
  float4 w2 = ld4(cw + (cp + 2) * 4);
  float4 w3 = ld4(cw + (cp + 3) * 4);
  float4 cbv = ld4(cb + cp);
  float4 acc;
  acc.x = cbv.x + w0.x * xs[0].x + w0.y * xs[1].x + w0.z * xs[2].x + w0.w * xs[3].x;
  acc.y = cbv.y + w1.x * xs[0].y + w1.y * xs[1].y + w1.z * xs[2].y + w1.w * xs[3].y;
  acc.z = cbv.z + w2.x * xs[0].z + w2.y * xs[1].z + w2.z * xs[2].z + w2.w * xs[3].z;
  acc.w = cbv.w + w3.x * xs[0].w + w3.y * xs[1].w + w3.z * xs[2].w + w3.w * xs[3].w;
  acc.x = acc.x / (1.f + __expf(-acc.x));
  acc.y = acc.y / (1.f + __expf(-acc.y));
  acc.z = acc.z / (1.f + __expf(-acc.z));
  acc.w = acc.w / (1.f + __expf(-acc.w));
  *(float4*)(xbca + bl * CONVDIM + cp) = acc;
}

// ---------------------------------------------------------------------------
// Kernel 3: per-chunk local state Hc[p,s] = sum_u exp(G-s_u) dt_u x_u[p] B_u[s]
// One block per (b,h,chunk). WMMA: (P=64) x (S=128), K=Q=64.
// ---------------------------------------------------------------------------
__global__ void k_chunk_state(const float* __restrict__ xbca,
                              const float* __restrict__ zx,
                              const float* __restrict__ dt_bias,
                              const float* __restrict__ A_log,
                              float* __restrict__ Hc,
                              float* __restrict__ sbuf,
                              float* __restrict__ gbuf) {
  __shared__ float sDt[Q], sS[Q];
  __shared__ __align__(16) __bf16 sW[HEADDIM * Q];   // A: 4 mtiles x 2 kblocks
  __shared__ __align__(16) __bf16 sBt[DSTATE * Q];   // B: 8 ntiles x 2 kblocks
  int tid = threadIdx.x;
  int c = blockIdx.x & 63;
  int h = (blockIdx.x >> 6) & 1;
  int b = blockIdx.x >> 7;
  int t0 = c * Q;
  int rowbase = b * SEQ + t0;
  int bh = b * NHEADS + h;
  if (tid < Q)
    sDt[tid] = softplusf(zx[(rowbase + tid) * ZXLD + 512 + h] + dt_bias[h]);
  __syncthreads();
  if (tid == 0) {
    float A = -__expf(A_log[h]);
    float run = 0.f;
    for (int u = 0; u < Q; ++u) { run += sDt[u] * A; sS[u] = run; }
  }
  __syncthreads();
  if (tid < Q) sbuf[bh * SEQ + t0 + tid] = sS[tid];
  if (tid == 0) gbuf[bh * NCHUNK + c] = sS[Q - 1];
  float G = sS[Q - 1];
  for (int e = tid; e < 1024; e += 256) {  // sW: vectorize over p (rows)
    int u = e & 63, p = (e >> 6) << 2;
    float wt = sDt[u] * __expf(G - sS[u]);
    float4 v = ld4(xbca + (rowbase + u) * CONVDIM + h * HEADDIM + p);
    v.x *= wt; v.y *= wt; v.z *= wt; v.w *= wt;
    stage4_rows(sW, p, u, 2, v);
  }
  for (int e = tid; e < 2048; e += 256) {  // sBt: vectorize over s (rows)
    int u = e >> 5, s = (e & 31) << 2;
    stage4_rows(sBt, s, u, 2, ld4(xbca + (rowbase + u) * CONVDIM + DINNER + s));
  }
  __syncthreads();
  int wave = tid >> 5, lane = tid & 31;
  int col = lane & 15, rb = (lane >> 4) ? 8 : 0;
  float* Hco = Hc + (size_t)(bh * NCHUNK + c) * PS;
  for (int tl = wave; tl < 32; tl += 8) {  // 4 mtiles x 8 ntiles
    int mt = tl >> 3, nt = tl & 7;
    v8f acc = zero8();
#pragma unroll
    for (int ks = 0; ks < 2; ++ks) {
      v16bf a = load_frag(sW + ((mt * 2 + ks) << 9));
      v16bf bb = load_frag(sBt + ((nt * 2 + ks) << 9));
      acc = wmma_bf16(a, bb, acc);
    }
#pragma unroll
    for (int i = 0; i < 8; ++i)
      Hco[(mt * 16 + rb + i) * DSTATE + nt * 16 + col] = acc[i];
  }
}

// ---------------------------------------------------------------------------
// Kernel 4: chunk-state recurrence (64 steps), float4 per thread over P*S.
// Stores the state *entering* each chunk.
// ---------------------------------------------------------------------------
__global__ void k_state_scan(const float* __restrict__ Hc,
                             const float* __restrict__ gbuf,
                             float* __restrict__ Hstate) {
  int bh = blockIdx.x >> 3;
  int pack = ((blockIdx.x & 7) * 256 + threadIdx.x) << 2;
  size_t base = (size_t)bh * NCHUNK * PS + pack;
  float4 h; h.x = h.y = h.z = h.w = 0.f;
  for (int c = 0; c < NCHUNK; ++c) {
    *(float4*)(Hstate + base + (size_t)c * PS) = h;
    float eg = __expf(gbuf[bh * NCHUNK + c]);
    float4 a = ld4(Hc + base + (size_t)c * PS);
    h.x = h.x * eg + a.x;
    h.y = h.y * eg + a.y;
    h.z = h.z * eg + a.z;
    h.w = h.w * eg + a.w;
  }
}

// ---------------------------------------------------------------------------
// Kernel 5: chunk outputs.
//   Gram = C @ B^T (64x64, K=128) -> masked decay M (bf16, fragment order)
//   Y    = exp(s)C @ H^T (K=128) + M @ (dt*X) (K=64) + D*x
// One block per (b,h,chunk). sC is rescaled by exp(s_t) in place after Gram.
// ---------------------------------------------------------------------------
__global__ void k_chunk_out(const float* __restrict__ xbca,
                            const float* __restrict__ zx,
                            const float* __restrict__ dt_bias,
                            const float* __restrict__ sbuf,
                            const float* __restrict__ Hstate,
                            const float* __restrict__ Dv,
                            float* __restrict__ y) {
  __shared__ float sS[Q], sE[Q], sDt[Q];
  __shared__ __align__(16) __bf16 sC[Q * DSTATE];   // A: 4 mt x 4 kb
  __shared__ __align__(16) __bf16 sBH[Q * DSTATE];  // B (N-major): 4 nt x 4 kb
  __shared__ __align__(16) __bf16 sX[Q * HEADDIM];  // B (N-major): 4 nt x 2 kb
  __shared__ __align__(16) __bf16 sM[Q * Q];        // A: 4 mt x 2 kb
  int tid = threadIdx.x;
  int c = blockIdx.x & 63;
  int h = (blockIdx.x >> 6) & 1;
  int b = blockIdx.x >> 7;
  int t0 = c * Q;
  int rowbase = b * SEQ + t0;
  int bh = b * NHEADS + h;
  if (tid < Q) {
    float sv = sbuf[bh * SEQ + t0 + tid];
    sS[tid] = sv;
    sE[tid] = __expf(sv);
    sDt[tid] = softplusf(zx[(rowbase + tid) * ZXLD + 512 + h] + dt_bias[h]);
  }
  __syncthreads();
  for (int e = tid; e < 2048; e += 256) {  // sC / sBH: K contiguous
    int t = e >> 5, sp = (e & 31) << 2;
    stage4(sC, t, sp, 4,
           ld4(xbca + (rowbase + t) * CONVDIM + DINNER + DSTATE + sp));
    stage4(sBH, t, sp, 4, ld4(xbca + (rowbase + t) * CONVDIM + DINNER + sp));
  }
  for (int e = tid; e < 1024; e += 256) {  // sX: vectorize over p (rows)
    int u = e & 63, p = (e >> 6) << 2;
    float4 v = ld4(xbca + (rowbase + u) * CONVDIM + h * HEADDIM + p);
    float dtu = sDt[u];
    v.x *= dtu; v.y *= dtu; v.z *= dtu; v.w *= dtu;
    stage4_rows(sX, p, u, 2, v);
  }
  __syncthreads();
  int wave = tid >> 5, lane = tid & 31;
  int col = lane & 15, rb = (lane >> 4) ? 8 : 0;
  // Gram + causal decay mask -> sM
  for (int tl = wave; tl < 16; tl += 8) {
    int mt = tl >> 2, nt = tl & 3;
    v8f acc = zero8();
#pragma unroll
    for (int ks = 0; ks < 4; ++ks) {
      v16bf a = load_frag(sC + ((mt * 4 + ks) << 9));
      v16bf bb = load_frag(sBH + ((nt * 4 + ks) << 9));
      acc = wmma_bf16(a, bb, acc);
    }
#pragma unroll
    for (int i = 0; i < 8; ++i) {
      int t = mt * 16 + rb + i, u = nt * 16 + col;
      float v = (t >= u) ? acc[i] * __expf(sS[t] - sS[u]) : 0.f;
      sM[shuf_idx(t, u, 2)] = (__bf16)v;
    }
  }
  __syncthreads();
  // Restage sBH with H[p][s] (B-operand: N=p, K=s); rescale sC by exp(s_t).
  const float* Hrow = Hstate + (size_t)(bh * NCHUNK + c) * PS;
  for (int e = tid; e < 2048; e += 256) {
    int p = e >> 5, sp = (e & 31) << 2;
    stage4(sBH, p, sp, 4, ld4(Hrow + p * DSTATE + sp));
    union { __bf16 h[4]; uint2 u; } pk;
    uint2* addr = (uint2*)(sC + shuf_idx(p, sp, 4));
    pk.u = *addr;
    float ep = sE[p];
    pk.h[0] = (__bf16)((float)pk.h[0] * ep);
    pk.h[1] = (__bf16)((float)pk.h[1] * ep);
    pk.h[2] = (__bf16)((float)pk.h[2] * ep);
    pk.h[3] = (__bf16)((float)pk.h[3] * ep);
    *addr = pk.u;
  }
  __syncthreads();
  float Dh = Dv[h];
  for (int tl = wave; tl < 16; tl += 8) {
    int mt = tl >> 2, nt = tl & 3;
    v8f acc = zero8();
#pragma unroll
    for (int ks = 0; ks < 4; ++ks) {  // inter-chunk: exp(s)C @ H^T
      v16bf a = load_frag(sC + ((mt * 4 + ks) << 9));
      v16bf bb = load_frag(sBH + ((nt * 4 + ks) << 9));
      acc = wmma_bf16(a, bb, acc);
    }
#pragma unroll
    for (int ks = 0; ks < 2; ++ks) {  // intra-chunk: M @ X
      v16bf a = load_frag(sM + ((mt * 2 + ks) << 9));
      v16bf bb = load_frag(sX + ((nt * 2 + ks) << 9));
      acc = wmma_bf16(a, bb, acc);
    }
#pragma unroll
    for (int i = 0; i < 8; ++i) {
      int tloc = mt * 16 + rb + i, p = nt * 16 + col;
      float xv = xbca[(rowbase + tloc) * CONVDIM + h * HEADDIM + p];
      y[(rowbase + tloc) * DINNER + h * HEADDIM + p] = acc[i] + Dh * xv;
    }
  }
}

// ---------------------------------------------------------------------------
// Kernel 6: gate (y * silu(z)) -> RMSNorm -> out_proj GEMM (K=128) + residual.
// Block = 64 rows, 8 waves.
// ---------------------------------------------------------------------------
__global__ void k_gate_out(const float* __restrict__ y,
                           const float* __restrict__ zx,
                           const float* __restrict__ nw,
                           const float* __restrict__ ow,
                           const float* __restrict__ xin,
                           float* __restrict__ xout) {
  __shared__ __align__(16) __bf16 sY[64 * DINNER];      // A: 4 mt x 4 kb
  __shared__ __align__(16) __bf16 sW[DMODEL * DINNER];  // B: 4 nt x 4 kb
  __shared__ float sScale[64];
  int tid = threadIdx.x;
  int m0 = blockIdx.x * 64;
  int r = tid >> 2, sub = tid & 3;
  float part = 0.f;
#pragma unroll
  for (int c8 = 0; c8 < 8; ++c8) {
    int col = sub * 32 + c8 * 4;
    float4 yv = ld4(y + (m0 + r) * DINNER + col);
    float4 zv = ld4(zx + (m0 + r) * ZXLD + col);
    float4 g;
    g.x = yv.x * (zv.x / (1.f + __expf(-zv.x)));
    g.y = yv.y * (zv.y / (1.f + __expf(-zv.y)));
    g.z = yv.z * (zv.z / (1.f + __expf(-zv.z)));
    g.w = yv.w * (zv.w / (1.f + __expf(-zv.w)));
    stage4(sY, r, col, 4, g);
    part += g.x * g.x + g.y * g.y + g.z * g.z + g.w * g.w;
  }
  part += __shfl_xor(part, 1, 32);
  part += __shfl_xor(part, 2, 32);
  if (sub == 0) sScale[r] = rsqrtf(part * (1.f / 128.f) + 1e-5f);
  for (int e = tid; e < 2048; e += 256) {  // sW: K contiguous
    int n = e >> 5, kp = (e & 31) << 2;
    stage4(sW, n, kp, 4, ld4(ow + n * DINNER + kp));
  }
  __syncthreads();
  float sc = sScale[r];
#pragma unroll
  for (int c8 = 0; c8 < 8; ++c8) {
    int col = sub * 32 + c8 * 4;
    union { __bf16 h[4]; uint2 u; } pk;
    uint2* addr = (uint2*)(sY + shuf_idx(r, col, 4));
    pk.u = *addr;
    pk.h[0] = (__bf16)((float)pk.h[0] * sc * nw[col + 0]);
    pk.h[1] = (__bf16)((float)pk.h[1] * sc * nw[col + 1]);
    pk.h[2] = (__bf16)((float)pk.h[2] * sc * nw[col + 2]);
    pk.h[3] = (__bf16)((float)pk.h[3] * sc * nw[col + 3]);
    *addr = pk.u;
  }
  __syncthreads();
  int wave = tid >> 5, lane = tid & 31;
  int col = lane & 15, rb = (lane >> 4) ? 8 : 0;
  for (int tl = wave; tl < 16; tl += 8) {
    int mt = tl >> 2, nt = tl & 3;
    v8f acc = zero8();
#pragma unroll
    for (int ks = 0; ks < 4; ++ks) {
      v16bf a = load_frag(sY + ((mt * 4 + ks) << 9));
      v16bf bb = load_frag(sW + ((nt * 4 + ks) << 9));
      acc = wmma_bf16(a, bb, acc);
    }
#pragma unroll
    for (int i = 0; i < 8; ++i) {
      int gm = m0 + mt * 16 + rb + i;
      int gc = nt * 16 + col;
      xout[gm * DMODEL + gc] = acc[i] + xin[gm * DMODEL + gc];
    }
  }
}

// ---------------------------------------------------------------------------
// Host launch
// ---------------------------------------------------------------------------
extern "C" void kernel_launch(void* const* d_in, const int* in_sizes, int n_in,
                              void* d_out, int out_size, void* d_ws,
                              size_t ws_size, hipStream_t stream) {
  (void)in_sizes; (void)n_in; (void)out_size; (void)ws_size;
  const float* xg        = (const float*)d_in[0];
  const float* in_proj_w = (const float*)d_in[1];
  const float* conv_w    = (const float*)d_in[2];
  const float* conv_b    = (const float*)d_in[3];
  const float* dt_bias   = (const float*)d_in[4];
  const float* A_log     = (const float*)d_in[5];
  const float* Dvals     = (const float*)d_in[6];
  const float* norm_w    = (const float*)d_in[7];
  const float* out_w     = (const float*)d_in[8];

  size_t off = 0;
  char* ws = (char*)d_ws;
  auto take = [&](size_t nfloat) -> float* {
    float* p = (float*)(ws + off);
    off += (nfloat * sizeof(float) + 255) & ~(size_t)255;
    return p;
  };
  float* zxb   = take((size_t)BL * ZXLD);
  float* xbca  = take((size_t)BL * CONVDIM);
  float* sbuf  = take((size_t)BSZ * NHEADS * SEQ);
  float* gbuf  = take((size_t)BSZ * NHEADS * NCHUNK);
  float* Hc    = take((size_t)BSZ * NHEADS * NCHUNK * PS);
  float* Hst   = take((size_t)BSZ * NHEADS * NCHUNK * PS);
  float* ybuf  = take((size_t)BL * DINNER);
  float* ping0 = take((size_t)BL * DMODEL);
  float* ping1 = take((size_t)BL * DMODEL);

  const float* xin = xg;
  for (int layer = 0; layer < N_LAYERS; ++layer) {
    float* xout = (layer == N_LAYERS - 1) ? (float*)d_out
                                          : ((layer & 1) ? ping1 : ping0);
    const float* iw  = in_proj_w + (size_t)layer * DINPROJ * DMODEL;
    const float* cw  = conv_w + (size_t)layer * CONVDIM * 4;
    const float* cb  = conv_b + (size_t)layer * CONVDIM;
    const float* dtb = dt_bias + (size_t)layer * NHEADS;
    const float* al  = A_log + (size_t)layer * NHEADS;
    const float* dp  = Dvals + (size_t)layer * NHEADS;
    const float* nw  = norm_w + (size_t)layer * DINNER;
    const float* ow  = out_w + (size_t)layer * DMODEL * DINNER;

    k_inproj<<<BL / 16, 256, 0, stream>>>(xin, iw, zxb);
    k_conv<<<(BL * 96) / 256, 256, 0, stream>>>(zxb, cw, cb, xbca);
    k_chunk_state<<<BSZ * NHEADS * NCHUNK, 256, 0, stream>>>(
        xbca, zxb, dtb, al, Hc, sbuf, gbuf);
    k_state_scan<<<BSZ * NHEADS * 8, 256, 0, stream>>>(Hc, gbuf, Hst);
    k_chunk_out<<<BSZ * NHEADS * NCHUNK, 256, 0, stream>>>(
        xbca, zxb, dtb, sbuf, Hst, dp, ybuf);
    k_gate_out<<<BL / 64, 256, 0, stream>>>(ybuf, zxb, nw, ow, xin, xout);
    xin = xout;
  }
}
